// GatedDeltaNet_4887672783152
// MI455X (gfx1250) — matrix operand order repaired and checked by reading, not compiled, not run on hardware
//
#include <hip/hip_runtime.h>
#include <hip/hip_bf16.h>

// ---------------------------------------------------------------------------
// Gated DeltaNet forward for MI455X (gfx1250, wave32, WMMA, async-LDS).
//
//   0) cvt_f32_bf16  : x, 7 weight matrices  -> bf16 once in HBM
//   1) gemm          : xp      = x_bf  @ in_proj^T   (f32 out, 16384x1024x1024)
//   2) conv_silu     : xpa_bf  = bf16(silu(conv4(xp)+b))
//   3-5) gemm        : q,k,v   = xpa_bf @ {W_q,W_k,W_v}^T   (f32 out)
//   6-7) gemm+sigmean: alpha,beta
//   8) l2norm q,k
//   9) scan          : 128 blocks (b,h), 64x64 state in VGPRs; cell_out -> bf16
//  10) gemm          : out = cell_bf @ out_proj^T -> d_out ; S_final from scan
//
// GEMM: 128x128 C tile, TK=64, bf16 operands in HBM; K-tiles staged to LDS
// with double-buffered GLOBAL_LOAD_ASYNC_TO_LDS_B128: issue tile i+1 into the
// alternate buffer, wait ASYNCcnt<=8 (in-order completion => tile i landed),
// then run 16 v_wmma_f32_16x16x32_bf16 on tile i while tile i+1 streams in.
// ---------------------------------------------------------------------------

#define BATCH 8
#define TLEN  2048
#define DMODEL 1024
#define NHEAD 16
#define HDIM  64

#define TM 128
#define TN 128
#define TK 64
#define LDK (TK + 8)          // 72 bf16 = 144 B row stride (16B-aligned, bank-spread)

typedef __attribute__((ext_vector_type(16))) __bf16        v16bf;
typedef __attribute__((ext_vector_type(8)))  float         v8f;
typedef __attribute__((ext_vector_type(4)))  unsigned int  uint4v;
typedef __attribute__((ext_vector_type(4)))  int           i4;

typedef __attribute__((address_space(1))) i4   as1_i4;   // __device__ int4
typedef __attribute__((address_space(3))) i4   as3_i4;   // __shared__ int4
typedef __attribute__((address_space(3))) void as3_void;

__device__ __forceinline__ unsigned short f2bf(float f) {
    union { float f; unsigned int u; } c; c.f = f;
    unsigned int u = c.u;
    u += 0x7FFFu + ((u >> 16) & 1u);   // round-to-nearest-even
    return (unsigned short)(u >> 16);
}

// 16-byte async copy global -> LDS (ASYNCcnt-tracked)
__device__ __forceinline__ void async_copy16(const unsigned short* g,
                                             unsigned short* l) {
#if __has_builtin(__builtin_amdgcn_global_load_async_to_lds_b128)
    __builtin_amdgcn_global_load_async_to_lds_b128(
        (as1_i4*)const_cast<unsigned short*>(g), (as3_i4*)l, 0, 0);
#else
    unsigned loff = (unsigned)(unsigned long long)(as3_void*)l;
    asm volatile("global_load_async_to_lds_b128 %0, %1, off"
                 :: "v"(loff), "v"((unsigned long long)g) : "memory");
#endif
}

__device__ __forceinline__ void wait_async0() {
#if __has_builtin(__builtin_amdgcn_s_wait_asynccnt)
    __builtin_amdgcn_s_wait_asynccnt(0);
#else
    asm volatile("s_wait_asynccnt 0x0" ::: "memory");
#endif
}

// previous 8-load group complete; the 8 just-issued may remain in flight
__device__ __forceinline__ void wait_async8() {
#if __has_builtin(__builtin_amdgcn_s_wait_asynccnt)
    __builtin_amdgcn_s_wait_asynccnt(8);
#else
    asm volatile("s_wait_asynccnt 0x8" ::: "memory");
#endif
}

// C[M,N] (f32) = A[M,K] (bf16, row-major) @ W[N,K]^T (bf16, row-major)
__global__ __launch_bounds__(256) void gemm_bf16_wmma(
    const unsigned short* __restrict__ A, const unsigned short* __restrict__ W,
    float* __restrict__ C, int M, int N, int K)
{
    __shared__ __align__(16) unsigned short lA[2][TM * LDK];
    __shared__ __align__(16) unsigned short lB[2][TN * LDK];

    const int tid  = threadIdx.x;
    const int lane = tid & 31;
    const int wave = tid >> 5;
    const int wr   = wave >> 2;      // 0..1 : 64-row strip of C
    const int wc   = wave & 3;       // 0..3 : 32-col strip of C
    const int m0   = blockIdx.y * TM;
    const int n0   = blockIdx.x * TN;

    // async staging: lane copies 16B (8 bf16); 8 lanes/row, 32 rows/issue, 4 issues
    const int cpRow = tid >> 3;      // 0..31
    const int cpSeg = tid & 7;       // 0..7

    const int laneM = lane & 15;
    const int kb0A  = (lane < 16) ? 0 : 8;    // ISA 16-bit A fragment K base
    const int kbB   = (lane < 16) ? 0 : 8;    // ISA 16-bit B fragment K base

    const unsigned short* gA = A + (size_t)(m0 + cpRow) * K + cpSeg * 8;
    const unsigned short* gW = W + (size_t)(n0 + cpRow) * K + cpSeg * 8;
    const unsigned lOff = cpRow * LDK + cpSeg * 8;

    v8f acc[4][2];
#pragma unroll
    for (int i = 0; i < 4; i++)
#pragma unroll
        for (int j = 0; j < 2; j++)
#pragma unroll
            for (int r = 0; r < 8; r++) acc[i][j][r] = 0.0f;

    union bfu { v16bf v; uint4v u[2]; };

    const int ntiles = K / TK;

    // prologue: stage tile 0 into buffer 0
#pragma unroll
    for (int c = 0; c < 4; c++) {
        async_copy16(gA + (size_t)(c * 32) * K,          &lA[0][lOff + c * 32 * LDK]);
        async_copy16(gW + (size_t)(c * 32) * K,          &lB[0][lOff + c * 32 * LDK]);
    }

    for (int t = 0; t < ntiles; t++) {
        const int cur = t & 1;
        if (t + 1 < ntiles) {
            const int nxt = cur ^ 1;
            const int kn  = (t + 1) * TK;
#pragma unroll
            for (int c = 0; c < 4; c++) {
                async_copy16(gA + (size_t)(c * 32) * K + kn, &lA[nxt][lOff + c * 32 * LDK]);
                async_copy16(gW + (size_t)(c * 32) * K + kn, &lB[nxt][lOff + c * 32 * LDK]);
            }
            wait_async8();   // tile t landed; tile t+1 still streaming
        } else {
            wait_async0();
        }
        __syncthreads();

#pragma unroll
        for (int kk = 0; kk < TK; kk += 32) {
            // B fragments: 2 x 8 bf16 per lane (k base, k base+16) -> 2 x ds_load_b128
            bfu bfrag[2];
#pragma unroll
            for (int j = 0; j < 2; j++) {
                const int nL = wc * 32 + j * 16 + laneM;
                const uint4v* p = (const uint4v*)&lB[cur][nL * LDK + kk + kbB];
                bfrag[j].u[0] = p[0];
                bfrag[j].u[1] = p[2];
            }
            // A fragments + WMMA
#pragma unroll
            for (int i = 0; i < 4; i++) {
                const int mL = wr * 64 + i * 16 + laneM;
                bfu ta;
                ta.u[0] = *(const uint4v*)&lA[cur][mL * LDK + kk + kb0A];
                ta.u[1] = *(const uint4v*)&lA[cur][mL * LDK + kk + kb0A + 16];
#pragma unroll
                for (int j = 0; j < 2; j++) {
                    acc[i][j] = __builtin_amdgcn_wmma_f32_16x16x32_bf16(
                        false, ta.v, false, bfrag[j].v,
                        (short)0, acc[i][j], false, false);
                }
            }
        }
        __syncthreads();   // readers of buf `cur` done before tile t+2 overwrites it
    }

    // ---- store C (D layout: VGPR r -> M = r (+8 for lanes 16-31)) ----
#pragma unroll
    for (int i = 0; i < 4; i++)
#pragma unroll
        for (int j = 0; j < 2; j++) {
            const int col   = n0 + wc * 32 + j * 16 + laneM;
            const int mbase = m0 + wr * 64 + i * 16 + ((lane < 16) ? 0 : 8);
#pragma unroll
            for (int r = 0; r < 8; r++)
                C[(size_t)(mbase + r) * N + col] = acc[i][j][r];
        }
}

// f32 -> bf16 buffer convert, 4 elems/thread
__global__ __launch_bounds__(256) void cvt_f32_bf16(
    const float* __restrict__ s, unsigned short* __restrict__ d)
{
    const int i = blockIdx.x * 256 + threadIdx.x;
    float4 f = ((const float4*)s)[i];
    uint2 u;
    u.x = ((unsigned)f2bf(f.y) << 16) | f2bf(f.x);
    u.y = ((unsigned)f2bf(f.w) << 16) | f2bf(f.z);
    ((uint2*)d)[i] = u;
}

// xpa[b,t,d] = bf16( silu( sum_j w[d,j]*xp[b,t-3+j,d] + bias[d] ) )
__global__ __launch_bounds__(256) void conv_silu_kernel(
    const float* __restrict__ xp, const float* __restrict__ w,
    const float* __restrict__ bias, unsigned short* __restrict__ out)
{
    const int idx = blockIdx.x * 256 + threadIdx.x;
    const int d = idx & (DMODEL - 1);
    const int t = (idx / DMODEL) & (TLEN - 1);
    const int b = idx / (DMODEL * TLEN);
    float acc = bias[d];
#pragma unroll
    for (int j = 0; j < 4; j++) {
        const int tt = t - 3 + j;
        if (tt >= 0)
            acc += w[d * 4 + j] * xp[((size_t)b * TLEN + tt) * DMODEL + d];
    }
    out[idx] = f2bf(acc / (1.0f + __expf(-acc)));
}

// per-(b,t,h): v / max(||v||, 1e-12) over 64 elems, in place (f32)
__global__ __launch_bounds__(256) void l2norm_kernel(float* __restrict__ data)
{
    const int idx = blockIdx.x * 256 + threadIdx.x;   // 0 .. B*T*H-1
    const int h  = idx & (NHEAD - 1);
    const int bt = idx / NHEAD;
    float4* p = (float4*)(data + (size_t)bt * DMODEL + h * HDIM);
    float ss = 0.0f;
    float4 vbuf[16];
#pragma unroll
    for (int i = 0; i < 16; i++) {
        float4 vv = p[i];
        vbuf[i] = vv;
        ss += vv.x * vv.x + vv.y * vv.y + vv.z * vv.z + vv.w * vv.w;
    }
    const float inv = 1.0f / fmaxf(sqrtf(ss), 1e-12f);
#pragma unroll
    for (int i = 0; i < 16; i++) {
        float4 vv = vbuf[i];
        vv.x *= inv; vv.y *= inv; vv.z *= inv; vv.w *= inv;
        p[i] = vv;
    }
}

// per-(b,t,h): mean_e sigmoid(pre[..,h*64+e] + bias[h*64+e])
__global__ __launch_bounds__(256) void sigmean_kernel(
    const float* __restrict__ pre, const float* __restrict__ bias,
    float* __restrict__ out)
{
    const int idx = blockIdx.x * 256 + threadIdx.x;   // 0 .. B*T*H-1
    const int h  = idx & (NHEAD - 1);
    const int bt = idx / NHEAD;
    const float* p  = pre + (size_t)bt * DMODEL + h * HDIM;
    const float* bb = bias + h * HDIM;
    float s = 0.0f;
#pragma unroll 8
    for (int e = 0; e < HDIM; e++) {
        const float x = p[e] + bb[e];
        s += 1.0f / (1.0f + __expf(-x));
    }
    out[idx] = s * (1.0f / HDIM);
}

// Recurrence: one block per (b,h); thread d owns state row S[d][0..63] in VGPRs.
__global__ __launch_bounds__(64) void scan_kernel(
    const float* __restrict__ q, const float* __restrict__ k,
    const float* __restrict__ v, const float* __restrict__ alpha,
    const float* __restrict__ beta, unsigned short* __restrict__ cell,
    float* __restrict__ Sfin)
{
    const int d  = threadIdx.x;
    const int bh = blockIdx.x;
    const int b  = bh / NHEAD;
    const int h  = bh % NHEAD;

    float S[HDIM];
#pragma unroll
    for (int e = 0; e < HDIM; e++) S[e] = 0.0f;

    __shared__ float kq[2 * HDIM];   // [0:64)=k_t, [64:128)=q_t

    for (int t = 0; t < TLEN; t++) {
        const size_t base  = ((size_t)b * TLEN + t) * DMODEL + h * HDIM;
        const size_t abase = (size_t)(b * TLEN + t) * NHEAD + h;
        kq[d]        = k[base + d];
        kq[HDIM + d] = q[base + d];
        const float vd = v[base + d];
        const float a  = alpha[abase];
        const float bt = beta[abase];
        __syncthreads();

        float Sk = 0.0f;
#pragma unroll
        for (int e = 0; e < HDIM; e++) Sk += S[e] * kq[e];

        const float bSk = bt * Sk;
        const float bv  = bt * vd;
        float outv = 0.0f;
#pragma unroll
        for (int e = 0; e < HDIM; e++) {
            const float ke = kq[e];
            const float s  = a * (S[e] - bSk * ke) + bv * ke;
            S[e] = s;
            outv += s * kq[HDIM + e];
        }
        cell[base + d] = f2bf(outv);
        __syncthreads();
    }

    float* dst = Sfin + ((size_t)bh * HDIM + d) * HDIM;
#pragma unroll
    for (int e = 0; e < HDIM; e++) dst[e] = S[e];
}

extern "C" void kernel_launch(void* const* d_in, const int* in_sizes, int n_in,
                              void* d_out, int out_size, void* d_ws, size_t ws_size,
                              hipStream_t stream) {
    const float* x         = (const float*)d_in[0];
    const float* in_proj_w = (const float*)d_in[1];
    const float* conv_w    = (const float*)d_in[2];
    const float* conv_b    = (const float*)d_in[3];
    const float* W_q       = (const float*)d_in[4];
    const float* W_k       = (const float*)d_in[5];
    const float* W_v       = (const float*)d_in[6];
    const float* W_alpha   = (const float*)d_in[7];
    const float* b_alpha   = (const float*)d_in[8];
    const float* W_beta    = (const float*)d_in[9];
    const float* b_beta    = (const float*)d_in[10];
    const float* out_proj  = (const float*)d_in[11];

    float* out_main = (float*)d_out;                                   // (B,T,DIM)
    float* out_S    = out_main + (size_t)BATCH * TLEN * DMODEL;        // (B,H,64,64)

    const size_t MT = (size_t)BATCH * TLEN;      // 16384
    const size_t WN = (size_t)DMODEL * DMODEL;   // 1M elems per weight

    // workspace carve-up (~400 MB)
    float* xp    = (float*)d_ws;                 // MT*1024 f32
    float* qb    = xp    + MT * DMODEL;
    float* kb    = qb    + MT * DMODEL;
    float* vb    = kb    + MT * DMODEL;
    float* pre   = vb    + MT * DMODEL;
    float* alpha = pre   + MT * DMODEL;
    float* beta  = alpha + MT * NHEAD;
    unsigned short* xbf  = (unsigned short*)(beta + MT * NHEAD);  // x bf16
    unsigned short* xpa  = xbf + MT * DMODEL;                     // act bf16
    unsigned short* wbf  = xpa + MT * DMODEL;                     // 7 weights bf16
    unsigned short* w_in = wbf + 0 * WN;
    unsigned short* w_q  = wbf + 1 * WN;
    unsigned short* w_k  = wbf + 2 * WN;
    unsigned short* w_v  = wbf + 3 * WN;
    unsigned short* w_a  = wbf + 4 * WN;
    unsigned short* w_b  = wbf + 5 * WN;
    unsigned short* w_o  = wbf + 6 * WN;
    unsigned short* cell = (unsigned short*)xp;  // xp dead after conv -> reuse

    const dim3 gGemm(DMODEL / TN, MT / TM);      // (8, 128)
    const dim3 bGemm(256);
    const int  gW = (int)(WN / 4 / 256);         // 1024 blocks per weight cvt
    const int  gX = (int)(MT * DMODEL / 4 / 256);

    // 0) one-time bf16 conversion of GEMM operands
    cvt_f32_bf16<<<gX, 256, 0, stream>>>(x, xbf);
    cvt_f32_bf16<<<gW, 256, 0, stream>>>(in_proj_w, w_in);
    cvt_f32_bf16<<<gW, 256, 0, stream>>>(W_q,     w_q);
    cvt_f32_bf16<<<gW, 256, 0, stream>>>(W_k,     w_k);
    cvt_f32_bf16<<<gW, 256, 0, stream>>>(W_v,     w_v);
    cvt_f32_bf16<<<gW, 256, 0, stream>>>(W_alpha, w_a);
    cvt_f32_bf16<<<gW, 256, 0, stream>>>(W_beta,  w_b);
    cvt_f32_bf16<<<gW, 256, 0, stream>>>(out_proj, w_o);

    // 1) xp = x @ in_proj_w^T
    gemm_bf16_wmma<<<gGemm, bGemm, 0, stream>>>(xbf, w_in, xp, (int)MT, DMODEL, DMODEL);
    // 2) depthwise conv + bias + silu -> bf16 activations
    conv_silu_kernel<<<(MT * DMODEL) / 256, 256, 0, stream>>>(xp, conv_w, conv_b, xpa);
    // 3-5) q,k,v projections
    gemm_bf16_wmma<<<gGemm, bGemm, 0, stream>>>(xpa, w_q, qb, (int)MT, DMODEL, DMODEL);
    gemm_bf16_wmma<<<gGemm, bGemm, 0, stream>>>(xpa, w_k, kb, (int)MT, DMODEL, DMODEL);
    gemm_bf16_wmma<<<gGemm, bGemm, 0, stream>>>(xpa, w_v, vb, (int)MT, DMODEL, DMODEL);
    // 6) alpha
    gemm_bf16_wmma<<<gGemm, bGemm, 0, stream>>>(xpa, w_a, pre, (int)MT, DMODEL, DMODEL);
    sigmean_kernel<<<(MT * NHEAD) / 256, 256, 0, stream>>>(pre, b_alpha, alpha);
    // 7) beta (reuses pre)
    gemm_bf16_wmma<<<gGemm, bGemm, 0, stream>>>(xpa, w_b, pre, (int)MT, DMODEL, DMODEL);
    sigmean_kernel<<<(MT * NHEAD) / 256, 256, 0, stream>>>(pre, b_beta, beta);
    // 8) l2-normalize q,k per head row
    l2norm_kernel<<<(MT * NHEAD) / 256, 256, 0, stream>>>(qb);
    l2norm_kernel<<<(MT * NHEAD) / 256, 256, 0, stream>>>(kb);
    // 9) recurrent scan; writes bf16 cell_out and f32 S_final
    scan_kernel<<<BATCH * NHEAD, HDIM, 0, stream>>>(qb, kb, vb, alpha, beta,
                                                    cell, out_S);
    // 10) output projection -> d_out
    gemm_bf16_wmma<<<gGemm, bGemm, 0, stream>>>(cell, w_o, out_main,
                                                (int)MT, DMODEL, DMODEL);
}
